// TemporalEigenstateNetwork_91311004713378
// MI455X (gfx1250) — matrix operand order repaired and checked by reading, not compiled, not run on hardware
//
#include <hip/hip_runtime.h>
#include <hip/hip_bf16.h>
#include <math.h>

typedef __attribute__((ext_vector_type(16))) __bf16 v16bf;
typedef __attribute__((ext_vector_type(8)))  __bf16 v8bf;
typedef __attribute__((ext_vector_type(8)))  float  v8f;
typedef __attribute__((ext_vector_type(4)))  float  v4f;
typedef __attribute__((ext_vector_type(4)))  int    v4i;

#define DT_C    0.1f
#define EPS_RES 0.01f
#define E_MINC  0.1f
#define E_CLIPC 0.99f
#define CHUNK 128
#define BB 8
#define TT 4096
#define DD 1024
#define KX 128
#define NCH (TT/CHUNK)   /* 32 */
#define BT  (BB*TT)      /* 32768 */

// --------------------------------------------------------------------------
// Async global->LDS support (gfx1250). Builtin signature (from compiler
// diagnostics): (int4 AS1*, int4 AS3*, imm offset, imm cpol).
// --------------------------------------------------------------------------
#define GLB_AS __attribute__((address_space(1)))
#define LDS_AS __attribute__((address_space(3)))

#if defined(__has_builtin)
#if __has_builtin(__builtin_amdgcn_global_load_async_to_lds_b128) && \
    __has_builtin(__builtin_amdgcn_s_wait_asynccnt)
#define HAVE_ASYNC_LDS 1
#endif
#endif

// Copy 64 contiguous bytes global -> LDS for this lane (4x b128 async ops).
__device__ __forceinline__ void lds_copy64(const void* g, void* l) {
#ifdef HAVE_ASYNC_LDS
  GLB_AS v4i* gp = (GLB_AS v4i*)g;
  LDS_AS v4i* lp = (LDS_AS v4i*)l;
#pragma unroll
  for (int j = 0; j < 4; ++j)
    __builtin_amdgcn_global_load_async_to_lds_b128(gp + j, lp + j, 0, 0);
#else
  const v4f* gs = (const v4f*)g;
  v4f*       ld = (v4f*)l;
#pragma unroll
  for (int j = 0; j < 4; ++j) ld[j] = gs[j];
#endif
}

__device__ __forceinline__ void lds_copy_wait() {
#ifdef HAVE_ASYNC_LDS
  __builtin_amdgcn_s_wait_asynccnt(0);
#endif
}

// ---------------------------------------------------------------------------
// Kernel 1: norm(M), R = I + eps*M/||M||_F, lambda = mag*e^{i*omega*dt},
//           power table lambda^p for p=0..CHUNK
// ---------------------------------------------------------------------------
__global__ void setup_kernel(const float* __restrict__ alpha_raw,
                             const float* __restrict__ omega,
                             const float* __restrict__ Mmat,
                             float* __restrict__ R,
                             float* __restrict__ lamr, float* __restrict__ lami,
                             float* __restrict__ Pr, float* __restrict__ Pi) {
  __shared__ float red[256];
  __shared__ float s_scale;
  int tid = threadIdx.x;
  float acc = 0.f;
  for (int i = tid; i < KX*KX; i += 256) { float v = Mmat[i]; acc += v*v; }
  red[tid] = acc; __syncthreads();
  for (int s = 128; s > 0; s >>= 1) {
    if (tid < s) red[tid] += red[tid+s];
    __syncthreads();
  }
  if (tid == 0) s_scale = EPS_RES / (sqrtf(red[0]) + 1e-8f);
  __syncthreads();
  float sc = s_scale;
  for (int i = tid; i < KX*KX; i += 256) {
    float v = Mmat[i] * sc;
    if ((i / KX) == (i % KX)) v += 1.0f;
    R[i] = v;
  }
  if (tid < KX) {
    float a   = alpha_raw[tid];
    float mag = E_MINC + (E_CLIPC - E_MINC) / (1.0f + expf(-a));
    float c = cosf(omega[tid]*DT_C), s = sinf(omega[tid]*DT_C);
    float lr = mag*c, li = mag*s;
    lamr[tid] = lr; lami[tid] = li;
    float pr = 1.f, pi = 0.f;
    Pr[tid] = pr; Pi[tid] = pi;
    for (int p = 1; p <= CHUNK; ++p) {
      float nr = pr*lr - pi*li;
      float ni = pr*li + pi*lr;
      pr = nr; pi = ni;
      Pr[p*KX + tid] = pr; Pi[p*KX + tid] = pi;
    }
  }
}

// ---------------------------------------------------------------------------
// Kernel 2: W_in (K x D fp32) -> transposed bf16 (D x K row-major)
// ---------------------------------------------------------------------------
__global__ void transpose_win_bf16_kernel(const float* __restrict__ W_in,
                                          __bf16* __restrict__ WinT) {
  int i = blockIdx.x*256 + threadIdx.x;    // i = d*KX + n
  int d = i >> 7, n = i & (KX-1);
  WinT[i] = (__bf16)W_in[(size_t)n*DD + d];
}

// ---------------------------------------------------------------------------
// Kernel 3: G = R @ W_out^T  (K x D, inner 128), stored bf16 row-major
// ---------------------------------------------------------------------------
__global__ void build_G_kernel(const float* __restrict__ R,
                               const float* __restrict__ W_out,
                               __bf16* __restrict__ Gbf) {
  int i = blockIdx.x*256 + threadIdx.x;   // i = k*DD + d
  int k = i >> 10, d = i & (DD-1);
  const float* rrow = R + k*KX;
  const float* wrow = W_out + (size_t)d*KX;
  float acc = 0.f;
#pragma unroll 8
  for (int j = 0; j < KX; ++j) acc = fmaf(rrow[j], wrow[j], acc);
  Gbf[i] = (__bf16)acc;
}

// ---------------------------------------------------------------------------
// Kernel 4/8: bf16 WMMA GEMM, C[M,N](f32) = A[M,K] * B[K,N], both row-major.
//   Fragments loaded straight from global per ISA 7.12.2 (table order: lane half h
//   holds K 8h..8h+7 and 16+8h..16+8h+7). Block = 8 waves; wave tile = 32x32 (2x2 WMMA
//   tiles); block tile = 128x64.
//   PRELOAD_B: the block's whole B slice (Ktot x 64, needs Ktot==128 -> 16 KB)
//   is staged once into LDS with async global->LDS copies; B frags then come
//   from strided 16-bit LDS reads.
// ---------------------------------------------------------------------------
template <int A_F32, int PRELOAD_B>
__global__ void __launch_bounds__(256)
gemm_wmma_direct(const void* __restrict__ Aglob, int lda,
                 const __bf16* __restrict__ Bglob, int ldb,
                 float* __restrict__ C, int ldc, int Ktot) {
  extern __shared__ __bf16 sB[];          // PRELOAD_B: Ktot x 64 bf16
  const int tid  = threadIdx.x;
  const int wave = tid >> 5;
  const int lane = tid & 31;
  const int nInBlk = (wave >> 2) * 32;                  // 0 or 32
  const int m0 = blockIdx.x * 128 + (wave & 3) * 32;    // wave's 32 rows
  const int n0 = blockIdx.y * 64 + nInBlk;              // wave's 32 cols

  const int rA    = lane & 15;          // A-frag row within 16
  const int kHalf = (lane >> 4) * 8;    // A-frag K sub-offset per lane half

  if (PRELOAD_B) {
    // 128 rows x 128B; each thread copies 64B (half a row) via async b128s
    int row = tid >> 1, half = tid & 1;
    const __bf16* gsrc = Bglob + (size_t)row*ldb + blockIdx.y*64 + half*32;
    __bf16*       ldst = sB + row*64 + half*32;
    lds_copy64(gsrc, ldst);
    lds_copy_wait();
    __syncthreads();
  }

  v8f acc[2][2];
#pragma unroll
  for (int a = 0; a < 2; ++a)
#pragma unroll
    for (int b = 0; b < 2; ++b) acc[a][b] = (v8f){};

  const float*  Af = (const float*)Aglob;
  const __bf16* Ah = (const __bf16*)Aglob;

  for (int k0 = 0; k0 < Ktot; k0 += 32) {
    // ---- A fragments: issue ALL loads first, then convert (overlap) --------
    v16bf a0, a1;
    if (A_F32) {
      const float* p0 = Af + (size_t)(m0 + rA)*lda      + k0 + kHalf;
      const float* p1 = Af + (size_t)(m0 + 16 + rA)*lda + k0 + kHalf;
      v4f f00 = *(const v4f*)(p0);
      v4f f01 = *(const v4f*)(p0 + 4);
      v4f f02 = *(const v4f*)(p0 + 16);
      v4f f03 = *(const v4f*)(p0 + 20);
      v4f f10 = *(const v4f*)(p1);
      v4f f11 = *(const v4f*)(p1 + 4);
      v4f f12 = *(const v4f*)(p1 + 16);
      v4f f13 = *(const v4f*)(p1 + 20);
#pragma unroll
      for (int j = 0; j < 4; ++j) {
        a0[j]    = (__bf16)f00[j];  a0[4+j]  = (__bf16)f01[j];
        a0[8+j]  = (__bf16)f02[j];  a0[12+j] = (__bf16)f03[j];
        a1[j]    = (__bf16)f10[j];  a1[4+j]  = (__bf16)f11[j];
        a1[8+j]  = (__bf16)f12[j];  a1[12+j] = (__bf16)f13[j];
      }
    } else {
      const __bf16* p0 = Ah + (size_t)(m0 + rA)*lda      + k0 + kHalf;
      const __bf16* p1 = Ah + (size_t)(m0 + 16 + rA)*lda + k0 + kHalf;
      v8bf l0 = *(const v8bf*)p0;
      v8bf h0 = *(const v8bf*)(p0 + 16);
      v8bf l1 = *(const v8bf*)p1;
      v8bf h1 = *(const v8bf*)(p1 + 16);
      a0 = __builtin_shufflevector(l0, h0, 0,1,2,3,4,5,6,7,8,9,10,11,12,13,14,15);
      a1 = __builtin_shufflevector(l1, h1, 0,1,2,3,4,5,6,7,8,9,10,11,12,13,14,15);
    }

    // ---- B fragments: lane%16 = N-col of tile, 16 K-elements in table order
    const int kBh = (lane >> 4) * 8;
    const __bf16* bp = PRELOAD_B ? (const __bf16*)(sB + (k0 + kBh)*64 + nInBlk + (lane & 15))
                                 : (Bglob + (size_t)(k0 + kBh)*ldb + n0 + (lane & 15));
    const int bstr = PRELOAD_B ? 64 : ldb;
    v16bf b0, b1;
#pragma unroll
    for (int j = 0; j < 16; ++j) {
      int kk = (j < 8) ? j : j + 8;
      b0[j] = bp[(size_t)kk*bstr];
      b1[j] = bp[(size_t)kk*bstr + 16];
    }

    acc[0][0] = __builtin_amdgcn_wmma_f32_16x16x32_bf16(false, a0, false, b0,
                  (short)0, acc[0][0], false, false);
    acc[0][1] = __builtin_amdgcn_wmma_f32_16x16x32_bf16(false, a0, false, b1,
                  (short)0, acc[0][1], false, false);
    acc[1][0] = __builtin_amdgcn_wmma_f32_16x16x32_bf16(false, a1, false, b0,
                  (short)0, acc[1][0], false, false);
    acc[1][1] = __builtin_amdgcn_wmma_f32_16x16x32_bf16(false, a1, false, b1,
                  (short)0, acc[1][1], false, false);

    // prefetch next K-tile of A into the near (WGP) cache level:
    // locality=3 -> scope 0 = pull into all cache levels (ISA 10.5)
    if (k0 + 32 < Ktot) {
      if (A_F32)
        __builtin_prefetch(Af + (size_t)(m0 + rA)*lda + k0 + 32, 0, 3);
      else
        __builtin_prefetch(Ah + (size_t)(m0 + rA)*lda + k0 + 32, 0, 3);
    }
  }

  // Epilogue: C/D layout = VGPR r: lanes 0-15 -> M=r, lanes 16-31 -> M=r+8
  const int colBase = n0 + (lane & 15);
  const int rowHalf = (lane >= 16) ? 8 : 0;
#pragma unroll
  for (int ms = 0; ms < 2; ++ms)
#pragma unroll
    for (int ns = 0; ns < 2; ++ns) {
      int grow0 = m0 + ms*16 + rowHalf;
      int col   = colBase + ns*16;
#pragma unroll
      for (int r = 0; r < 8; ++r)
        C[(size_t)(grow0 + r)*ldc + col] = acc[ms][ns][r];
    }
}

// ---------------------------------------------------------------------------
// Kernel 5: chunk-local zero-carry scans. 32768 independent 128-step scans.
// ---------------------------------------------------------------------------
__global__ void scan_kernel(const float* __restrict__ beta,
                            const float* __restrict__ lamr,
                            const float* __restrict__ lami,
                            float* __restrict__ slocr,
                            float* __restrict__ dlr, float* __restrict__ dli) {
  int tid = blockIdx.x*256 + threadIdx.x;   // NCH*BB*KX = 32768
  int k = tid & (KX-1);
  int b = (tid >> 7) & (BB-1);
  int c = tid >> 10;
  float lr = lamr[k], li = lami[k];
  float r = 0.f, im = 0.f;
  size_t base = ((size_t)b*TT + (size_t)c*CHUNK) * KX + k;
  for (int t = 0; t < CHUNK; ++t) {
    float bv = beta[base + (size_t)t*KX];
    float nr = fmaf(lr, r, fmaf(-li, im, bv));
    float ni = fmaf(li, r, lr*im);
    r = nr; im = ni;
    slocr[base + (size_t)t*KX] = r;
  }
  int di = (c*BB + b)*KX + k;
  dlr[di] = r; dli[di] = im;
}

// ---------------------------------------------------------------------------
// Kernel 6: cross-chunk carry recurrence (32 sequential steps, one workgroup)
//   h_{c+1} = (Lambda^128 * h_c + dlast_c) @ R. R (64 KB) is staged into LDS
//   once (async global->LDS) since it is re-read 32 times.
// ---------------------------------------------------------------------------
__global__ void __launch_bounds__(1024)
carry_kernel(const float* __restrict__ dlr, const float* __restrict__ dli,
             const float* __restrict__ Pr, const float* __restrict__ Pi,
             const float* __restrict__ Rg,
             float* __restrict__ Hr, float* __restrict__ Hi) {
  __shared__ float sR[KX*KX];      // 64 KB
  __shared__ float str[BB*KX];
  __shared__ float sti[BB*KX];
  int tid = threadIdx.x;           // 1024 = BB*KX
  // stage R: each thread copies 64B contiguous
  lds_copy64(Rg + tid*16, sR + tid*16);
  lds_copy_wait();
  int k = tid & (KX-1);
  int b = tid >> 7;
  float p128r = Pr[CHUNK*KX + k], p128i = Pi[CHUNK*KX + k];
  float hr = 0.f, hi = 0.f;
  __syncthreads();
  for (int c = 0; c < NCH; ++c) {
    int idx = (c*BB + b)*KX + k;
    Hr[idx] = hr; Hi[idx] = hi;                      // carry INTO chunk c
    float tr = fmaf(p128r, hr, fmaf(-p128i, hi, dlr[idx]));
    float ti = fmaf(p128i, hr, fmaf( p128r, hi, dli[idx]));
    str[b*KX + k] = tr; sti[b*KX + k] = ti;
    __syncthreads();
    float nr = 0.f, ni = 0.f;
#pragma unroll 8
    for (int j = 0; j < KX; ++j) {
      float rv = sR[j*KX + k];
      nr = fmaf(str[b*KX + j], rv, nr);
      ni = fmaf(sti[b*KX + j], rv, ni);
    }
    hr = nr; hi = ni;
    __syncthreads();
  }
}

// ---------------------------------------------------------------------------
// Kernel 7: S_re[bt,k] = sloc_r + Re(Lambda^{t+1} * h_c), emitted as bf16 for
//           the fused output GEMM  out = S_re @ G.
// ---------------------------------------------------------------------------
__global__ void combine_kernel(const float* __restrict__ slocr,
                               const float* __restrict__ Hr,
                               const float* __restrict__ Hi,
                               const float* __restrict__ Pr,
                               const float* __restrict__ Pi,
                               __bf16* __restrict__ Sre) {
  int flat = blockIdx.x*256 + threadIdx.x;   // BT*KX total
  int k    = flat & (KX-1);
  int bt   = flat >> 7;
  int b    = bt >> 12;            // TT = 4096
  int trem = bt & (TT-1);
  int c    = trem >> 7;
  int t    = trem & (CHUNK-1);
  int hidx = (c*BB + b)*KX + k;
  float pr = Pr[(t+1)*KX + k], pi = Pi[(t+1)*KX + k];
  float v  = slocr[flat] + pr*Hr[hidx] - pi*Hi[hidx];
  Sre[flat] = (__bf16)v;
}

// ---------------------------------------------------------------------------
extern "C" void kernel_launch(void* const* d_in, const int* in_sizes, int n_in,
                              void* d_out, int out_size, void* d_ws, size_t ws_size,
                              hipStream_t stream) {
  const float* x     = (const float*)d_in[0];
  const float* alpha = (const float*)d_in[1];
  const float* omega = (const float*)d_in[2];
  const float* W_in  = (const float*)d_in[3];
  const float* W_out = (const float*)d_in[4];
  const float* Mmat  = (const float*)d_in[5];
  float* out = (float*)d_out;

  char* ws = (char*)d_ws;
  size_t off = 0;
  auto alloc = [&](size_t bytes) -> char* {
    char* p = ws + off;
    off += (bytes + 255) & ~(size_t)255;
    return p;
  };

  float*  R     = (float*) alloc((size_t)KX*KX*4);          // 64 KB
  float*  lamr  = (float*) alloc(KX*4);
  float*  lami  = (float*) alloc(KX*4);
  float*  Pr    = (float*) alloc((size_t)(CHUNK+1)*KX*4);   // 66 KB
  float*  Pi    = (float*) alloc((size_t)(CHUNK+1)*KX*4);
  __bf16* WinT  = (__bf16*)alloc((size_t)KX*DD*2);          // 256 KB (D x K)
  __bf16* Gbf   = (__bf16*)alloc((size_t)KX*DD*2);          // 256 KB (K x D)
  float*  beta  = (float*) alloc((size_t)BT*KX*4);          // 16 MB
  float*  slocr = (float*) alloc((size_t)BT*KX*4);          // 16 MB
  float*  dlr   = (float*) alloc((size_t)NCH*BB*KX*4);
  float*  dli   = (float*) alloc((size_t)NCH*BB*KX*4);
  float*  Hr    = (float*) alloc((size_t)NCH*BB*KX*4);
  float*  Hi    = (float*) alloc((size_t)NCH*BB*KX*4);
  // beta is fully consumed by scan_kernel before combine writes Sre -> alias.
  __bf16* Sre   = (__bf16*)beta;                            // 8 MB inside beta

  // 1. setup: norm, R, lambda, power table
  setup_kernel<<<1, 256, 0, stream>>>(alpha, omega, Mmat, R, lamr, lami, Pr, Pi);
  // 2. W_in -> bf16 transposed (D x K row-major)
  transpose_win_bf16_kernel<<<(KX*DD)/256, 256, 0, stream>>>(W_in, WinT);
  // 3. G = R @ W_out^T (bf16, K x D row-major)
  build_G_kernel<<<(KX*DD)/256, 256, 0, stream>>>(R, W_out, Gbf);
  // 4. beta = x @ W_in^T     (M=32768, N=128, K=1024)  fp32-A bf16 WMMA
  gemm_wmma_direct<1,0><<<dim3(BT/128, KX/64), 256, 0, stream>>>(
      x, DD, WinT, KX, beta, KX, DD);
  // 5. chunk-local scans
  scan_kernel<<<(NCH*BB*KX)/256, 256, 0, stream>>>(beta, lamr, lami, slocr, dlr, dli);
  // 6. cross-chunk carry recurrence
  carry_kernel<<<1, 1024, 0, stream>>>(dlr, dli, Pr, Pi, R, Hr, Hi);
  // 7. combine -> S_re (bf16)
  combine_kernel<<<((size_t)BT*KX)/256, 256, 0, stream>>>(slocr, Hr, Hi, Pr, Pi, Sre);
  // 8. out = S_re @ G        (M=32768, N=1024, K=128)  bf16 WMMA, B in LDS
  gemm_wmma_direct<0,1><<<dim3(BT/128, DD/64), 256, KX*64*2, stream>>>(
      Sre, KX, Gbf, DD, out, DD, KX);
}
